// ContextQueryAttention_9028021256202
// MI455X (gfx1250) — matrix operand order repaired and checked by reading, not compiled, not run on hardware
//
#include <hip/hip_runtime.h>

typedef __attribute__((ext_vector_type(16))) __bf16          v16bf;
typedef __attribute__((ext_vector_type(16))) unsigned short  v16u;
typedef __attribute__((ext_vector_type(8)))  float           v8f;

union BfVec { v16u u; v16bf b; };

__device__ __forceinline__ unsigned short f2bf(float x) {
  union { float f; unsigned u; } v; v.f = x;
  unsigned r = v.u + 0x7FFFu + ((v.u >> 16) & 1u);   // round-to-nearest-even
  return (unsigned short)(r >> 16);
}

// A-operand (16x32 bf16, MxK) from LDS row-major [m][k], pitch in elements.
__device__ __forceinline__ v16bf ldA(const unsigned short* base, int pitch,
                                     int row0, int k0, int lane) {
  int m  = lane & 15;
  int kb = k0 + ((lane >> 4) << 4);
  const unsigned short* p = base + (row0 + m) * pitch + kb;
  BfVec r;
#pragma unroll
  for (int t = 0; t < 16; ++t) r.u[t] = p[t];
  return r.b;
}

// B-operand (32x16 bf16, KxN) from LDS n-major [n][k], pitch in elements.
__device__ __forceinline__ v16bf ldB(const unsigned short* base, int pitch,
                                     int n0, int k0, int lane) {
  int n  = lane & 15;
  int kb = k0 + ((lane >> 4) << 4);
  const unsigned short* p = base + (n0 + n) * pitch + kb;
  BfVec r;
#pragma unroll
  for (int t = 0; t < 16; ++t) r.u[t] = p[t];
  return r.b;
}

__device__ __forceinline__ v8f wmma_bf16(v16bf a, v16bf b, v8f c) {
  return __builtin_amdgcn_wmma_f32_16x16x32_bf16(false, a, false, b,
                                                 (short)0, c, false, false);
}

#define Bn 32
#define LCn 2048
#define LQn 256
#define Dn 128

// ---------------------------------------------------------------------------
// Kernel 1: column pass. One WG per (batch, 64-wide j tile). 256 threads.
// Streams c in 64-row blocks; online column softmax over i; accumulates
// T[j][d] = sum_i softmax_i(r_i + X_ij) * c[i][d]; writes T (f32) to ws.
// ---------------------------------------------------------------------------
__global__ __launch_bounds__(256) void cqa_colpass(
    const float* __restrict__ c, const float* __restrict__ q,
    const float* __restrict__ W0, float* __restrict__ T_ws) {
  constexpr int IB = 64, JT = 64;
  constexpr int PD = Dn + 4;    // 132: pitch for [rows][d]
  constexpr int PI = IB + 4;    // 68:  pitch for [x][i]
  // LDS layout (bytes)
  constexpr int OFF_Q  = 0;                       // ushort [JT][PD] = 16896
  constexpr int OFF_CA = OFF_Q  + JT * PD * 2;    // ushort [IB][PD] = 16896
  constexpr int OFF_CT = OFF_CA + IB * PD * 2;    // ushort [Dn][PI] = 17408
  constexpr int OFF_PT = OFF_CT + Dn * PI * 2;    // ushort [JT][PI] = 8704
  constexpr int OFF_W1 = OFF_PT + JT * PI * 2;    // float  [Dn]
  constexpr int OFF_R  = OFF_W1 + Dn * 4;         // float  [IB]
  constexpr int OFF_RP = OFF_R  + IB * 4;         // float  [IB][4]
  constexpr int OFF_MR = OFF_RP + IB * 4 * 4;     // float  [JT]
  constexpr int OFF_ZR = OFF_MR + JT * 4;
  constexpr int OFF_FC = OFF_ZR + JT * 4;
  constexpr int OFF_MN = OFF_FC + JT * 4;
  constexpr int OFF_CP = OFF_MN + JT * 4;         // float  [4][JT]
  constexpr int SMEM1  = OFF_CP + 4 * JT * 4;     // 63744 bytes

  __shared__ __align__(16) unsigned char smem[SMEM1];
  unsigned short* qB = (unsigned short*)(smem + OFF_Q);
  unsigned short* cA = (unsigned short*)(smem + OFF_CA);
  unsigned short* cT = (unsigned short*)(smem + OFF_CT);
  unsigned short* PT = (unsigned short*)(smem + OFF_PT);
  float* w1s  = (float*)(smem + OFF_W1);
  float* r_s  = (float*)(smem + OFF_R);
  float* rp   = (float*)(smem + OFF_RP);
  float* mrun = (float*)(smem + OFF_MR);
  float* zrun = (float*)(smem + OFF_ZR);
  float* facs = (float*)(smem + OFF_FC);
  float* mnew = (float*)(smem + OFF_MN);
  float* colp = (float*)(smem + OFF_CP);

  const int b    = blockIdx.y;
  const int j0   = blockIdx.x * JT;
  const int t    = threadIdx.x;
  const int lane = t & 31;
  const int w    = t >> 5;        // 0..7
  const int half = lane >> 4;
  const int n    = lane & 15;
  const v8f zero = {0.f, 0.f, 0.f, 0.f, 0.f, 0.f, 0.f, 0.f};

  if (t < Dn) w1s[t] = W0[t];
  if (t < JT) { mrun[t] = -1e30f; zrun[t] = 0.f; }
  // Stage q tile once, with w3 folded in (B-operand of the X gemm).
  {
    int row = t >> 2, g = t & 3;
    const float* qp = q + ((size_t)b * LQn + j0 + row) * Dn + g * 32;
    unsigned short* dq = qB + row * PD + g * 32;
#pragma unroll
    for (int k2 = 0; k2 < 32; ++k2)
      dq[k2] = f2bf(qp[k2] * W0[2 * Dn + g * 32 + k2]);
  }
  __syncthreads();

  // Wave roles.
  const int wit = w >> 1;         // X: i tile 0..3
  const int wj0 = (w & 1) * 32;   // X: j base (two 16-wide tiles)
  const int tjt = w & 3;          // T: j tile 0..3
  const int td0 = (w >> 2) * 64;  // T: d base (four 16-wide tiles)

  v8f Tacc[4] = {zero, zero, zero, zero};

  for (int ib = 0; ib < LCn / IB; ++ib) {
    const int i0 = ib * IB;
    // Prefetch next c block (global_prefetch_b8).
    if (ib + 1 < LCn / IB)
      __builtin_prefetch(c + ((size_t)b * LCn + i0 + IB + (t >> 2)) * Dn + (t & 3) * 32, 0, 1);
    // Stage c block: cA = bf16(c) [i][d], cT = bf16(c) [d][i], r partials.
    {
      int row = t >> 2, g = t & 3;
      const float* cp = c + ((size_t)b * LCn + i0 + row) * Dn + g * 32;
      unsigned short* da = cA + row * PD + g * 32;
      float partial = 0.f;
#pragma unroll
      for (int k2 = 0; k2 < 32; ++k2) {
        float v = cp[k2];
        unsigned short h = f2bf(v);
        da[k2] = h;
        cT[(g * 32 + k2) * PI + row] = h;
        partial += v * w1s[g * 32 + k2];
      }
      rp[row * 4 + g] = partial;
    }
    __syncthreads();
    if (t < IB) r_s[t] = rp[t * 4] + rp[t * 4 + 1] + rp[t * 4 + 2] + rp[t * 4 + 3];
    __syncthreads();

    // X = c @ (q*w3)^T for this block: 2 j-tiles per wave, K = 128.
    v8f X[2] = {zero, zero};
#pragma unroll
    for (int ks = 0; ks < 4; ++ks) {
      v16bf a = ldA(cA, PD, wit * 16, ks * 32, lane);
#pragma unroll
      for (int jt = 0; jt < 2; ++jt) {
        v16bf bb = ldB(qB, PD, wj0 + jt * 16, ks * 32, lane);
        X[jt] = wmma_bf16(a, bb, X[jt]);
      }
    }
    // Column (over i) max of S = X + r[i].
#pragma unroll
    for (int jt = 0; jt < 2; ++jt) {
      float cm = -1e30f;
#pragma unroll
      for (int r8 = 0; r8 < 8; ++r8) {
        float v = X[jt][r8] + r_s[wit * 16 + r8 + 8 * half];
        X[jt][r8] = v;
        cm = fmaxf(cm, v);
      }
      cm = fmaxf(cm, __shfl_xor(cm, 16, 32));
      if (lane < 16) colp[wit * JT + wj0 + jt * 16 + n] = cm;
    }
    __syncthreads();
    if (t < JT) {
      float mb = fmaxf(fmaxf(colp[t], colp[JT + t]),
                       fmaxf(colp[2 * JT + t], colp[3 * JT + t]));
      float mo = mrun[t];
      float mn = fmaxf(mo, mb);
      mnew[t] = mn;
      facs[t] = __expf(mo - mn);
      mrun[t] = mn;
    }
    __syncthreads();
    // P = exp(S - m_new); write P^T (A-operand of T gemm); column sums.
#pragma unroll
    for (int jt = 0; jt < 2; ++jt) {
      int j = wj0 + jt * 16 + n;
      float mn = mnew[j];
      float cs = 0.f;
#pragma unroll
      for (int r8 = 0; r8 < 8; ++r8) {
        float p = __expf(X[jt][r8] - mn);
        cs += p;
        PT[j * PI + wit * 16 + r8 + 8 * half] = f2bf(p);
      }
      cs += __shfl_xor(cs, 16, 32);
      if (lane < 16) colp[wit * JT + j] = cs;
    }
    // Rescale running T accumulators.
#pragma unroll
    for (int dt = 0; dt < 4; ++dt)
#pragma unroll
      for (int r8 = 0; r8 < 8; ++r8)
        Tacc[dt][r8] *= facs[tjt * 16 + r8 + 8 * half];
    __syncthreads();
    if (t < JT)
      zrun[t] = zrun[t] * facs[t] +
                (colp[t] + colp[JT + t] + colp[2 * JT + t] + colp[3 * JT + t]);
    // T += P^T @ c : K = 64.
#pragma unroll
    for (int ks = 0; ks < 2; ++ks) {
      v16bf a = ldA(PT, PI, tjt * 16, ks * 32, lane);
#pragma unroll
      for (int dt = 0; dt < 4; ++dt) {
        v16bf bb = ldB(cT, PI, td0 + dt * 16, ks * 32, lane);
        Tacc[dt] = wmma_bf16(a, bb, Tacc[dt]);
      }
    }
    __syncthreads();
  }

  // Finalize: T = Tacc / Z, write f32 to workspace.
#pragma unroll
  for (int dt = 0; dt < 4; ++dt)
#pragma unroll
    for (int r8 = 0; r8 < 8; ++r8) {
      int jl = tjt * 16 + r8 + 8 * half;
      int d  = td0 + dt * 16 + n;
      T_ws[((size_t)b * LQn + j0 + jl) * Dn + d] = Tacc[dt][r8] / zrun[jl];
    }
}

// ---------------------------------------------------------------------------
// Kernel 2: row pass. One WG per (batch, 64-row i tile). 128 threads, 4 waves.
// X = (c*w3) @ q^T, in-register row softmax -> S1 (bf16 in LDS),
// A = S1 @ q, Bm = S1 @ T, out = [c, A, c*A, c*Bm].
// ---------------------------------------------------------------------------
__global__ __launch_bounds__(128) void cqa_rowpass(
    const float* __restrict__ c, const float* __restrict__ q,
    const float* __restrict__ W0, const float* __restrict__ T_ws,
    float* __restrict__ out) {
  constexpr int IB = 64;
  constexpr int PD = Dn + 4;    // 132
  constexpr int PJ = LQn + 4;   // 260 (S1 pitch)
  constexpr int PC = 32 + 4;    // 36  (qT/TT pitch, 32-wide j chunk)
  constexpr int OFF_CA = 0;                        // ushort [IB][PD] = 16896
  constexpr int OFF_QB = OFF_CA + IB * PD * 2;     // ushort [64][PD] = 16896
  constexpr int OFF_S1 = 0;                        // ushort [IB][PJ] = 33280 (overlays CA/QB)
  constexpr int OFF_QT = OFF_QB + 64 * PD * 2;     // ushort [Dn][PC] = 9216
  constexpr int OFF_TT = OFF_QT + Dn * PC * 2;     // ushort [Dn][PC] = 9216
  constexpr int OFF_S  = OFF_TT + Dn * PC * 2;     // float  [LQn]
  constexpr int OFF_W2 = OFF_S  + LQn * 4;         // float  [Dn]
  constexpr int OFF_W3 = OFF_W2 + Dn * 4;          // float  [Dn]
  constexpr int OFF_SP = OFF_W3 + Dn * 4;          // float  [64][2]
  constexpr int SMEM2  = OFF_SP + 64 * 2 * 4;      // 54784 bytes

  __shared__ __align__(16) unsigned char smem[SMEM2];
  unsigned short* cA = (unsigned short*)(smem + OFF_CA);
  unsigned short* qB = (unsigned short*)(smem + OFF_QB);
  unsigned short* S1 = (unsigned short*)(smem + OFF_S1);
  unsigned short* qT = (unsigned short*)(smem + OFF_QT);
  unsigned short* TT = (unsigned short*)(smem + OFF_TT);
  float* s_s = (float*)(smem + OFF_S);
  float* w2s = (float*)(smem + OFF_W2);
  float* w3s = (float*)(smem + OFF_W3);
  float* sp  = (float*)(smem + OFF_SP);

  const int b    = blockIdx.y;
  const int i0   = blockIdx.x * IB;
  const int t    = threadIdx.x;
  const int lane = t & 31;
  const int w    = t >> 5;        // 0..3 : 16-row strip per wave
  const int half = lane >> 4;
  const int n    = lane & 15;
  const v8f zero = {0.f, 0.f, 0.f, 0.f, 0.f, 0.f, 0.f, 0.f};

  if (t < Dn) { w2s[t] = W0[Dn + t]; w3s[t] = W0[2 * Dn + t]; }
  __syncthreads();
  // Stage cA = bf16(c * w3).
  {
    int row = t >> 1, g = t & 1;
    const float* cp = c + ((size_t)b * LCn + i0 + row) * Dn + g * 64;
    unsigned short* da = cA + row * PD + g * 64;
#pragma unroll
    for (int k2 = 0; k2 < 64; ++k2) da[k2] = f2bf(cp[k2] * w3s[g * 64 + k2]);
  }

  // ---- Phase 1: X = (c*w3) @ q^T over 4 chunks of 64 j ----
  v8f accX[16];
#pragma unroll
  for (int i = 0; i < 16; ++i) accX[i] = zero;

#pragma unroll
  for (int jc = 0; jc < 4; ++jc) {
    __syncthreads();
    {
      int row = t >> 1, g = t & 1;
      const float* qp = q + ((size_t)b * LQn + jc * 64 + row) * Dn + g * 64;
      unsigned short* dq = qB + row * PD + g * 64;
      float partial = 0.f;
#pragma unroll
      for (int k2 = 0; k2 < 64; ++k2) {
        float v = qp[k2];
        dq[k2] = f2bf(v);
        partial += v * w2s[g * 64 + k2];
      }
      sp[row * 2 + g] = partial;
    }
    __syncthreads();
    if (t < 64) s_s[jc * 64 + t] = sp[t * 2] + sp[t * 2 + 1];
#pragma unroll
    for (int ks = 0; ks < 4; ++ks) {
      v16bf a = ldA(cA, PD, w * 16, ks * 32, lane);
#pragma unroll
      for (int jt = 0; jt < 4; ++jt) {
        v16bf bb = ldB(qB, PD, jt * 16, ks * 32, lane);
        accX[jc * 4 + jt] = wmma_bf16(a, bb, accX[jc * 4 + jt]);
      }
    }
  }
  __syncthreads();   // all waves done with cA/qB; s_s complete

  // ---- Phase 1.5: row softmax fully in registers (wave32 shfl_xor) ----
  float rowmax[8], rowsum[8];
#pragma unroll
  for (int r8 = 0; r8 < 8; ++r8) rowmax[r8] = -1e30f;
#pragma unroll
  for (int tt = 0; tt < 16; ++tt) {
    float sj = s_s[tt * 16 + n];
#pragma unroll
    for (int r8 = 0; r8 < 8; ++r8) {
      float v = accX[tt][r8] + sj;
      accX[tt][r8] = v;
      rowmax[r8] = fmaxf(rowmax[r8], v);
    }
  }
#pragma unroll
  for (int r8 = 0; r8 < 8; ++r8) {
    float m = rowmax[r8];
    m = fmaxf(m, __shfl_xor(m, 1, 32));
    m = fmaxf(m, __shfl_xor(m, 2, 32));
    m = fmaxf(m, __shfl_xor(m, 4, 32));
    m = fmaxf(m, __shfl_xor(m, 8, 32));
    rowmax[r8] = m;
    rowsum[r8] = 0.f;
  }
#pragma unroll
  for (int tt = 0; tt < 16; ++tt)
#pragma unroll
    for (int r8 = 0; r8 < 8; ++r8) {
      float p = __expf(accX[tt][r8] - rowmax[r8]);
      accX[tt][r8] = p;
      rowsum[r8] += p;
    }
#pragma unroll
  for (int r8 = 0; r8 < 8; ++r8) {
    float s = rowsum[r8];
    s += __shfl_xor(s, 1, 32);
    s += __shfl_xor(s, 2, 32);
    s += __shfl_xor(s, 4, 32);
    s += __shfl_xor(s, 8, 32);
    rowsum[r8] = 1.f / s;
  }
  // Store S1 (bf16, A-operand layout). Each wave only re-reads its own rows.
#pragma unroll
  for (int tt = 0; tt < 16; ++tt)
#pragma unroll
    for (int r8 = 0; r8 < 8; ++r8) {
      int row = w * 16 + r8 + 8 * half;
      S1[row * PJ + tt * 16 + n] = f2bf(accX[tt][r8] * rowsum[r8]);
    }

  // ---- Phase 2: A = S1 @ q and Bm = S1 @ T over 8 chunks of 32 j ----
  v8f accA[8], accB[8];
#pragma unroll
  for (int i = 0; i < 8; ++i) { accA[i] = zero; accB[i] = zero; }

  for (int jc = 0; jc < 8; ++jc) {
    __syncthreads();
    {
      // Transposed staging: thread t owns d-row t; coalesced global reads.
      const float* qp = q + ((size_t)b * LQn + jc * 32) * Dn + t;
      const float* tp = T_ws + ((size_t)b * LQn + jc * 32) * Dn + t;
#pragma unroll
      for (int jj = 0; jj < 32; ++jj) {
        qT[t * PC + jj] = f2bf(qp[(size_t)jj * Dn]);
        TT[t * PC + jj] = f2bf(tp[(size_t)jj * Dn]);
      }
    }
    __syncthreads();
    v16bf a = ldA(S1, PJ, w * 16, jc * 32, lane);
#pragma unroll
    for (int dt = 0; dt < 8; ++dt) {
      v16bf bq = ldB(qT, PC, dt * 16, 0, lane);
      accA[dt] = wmma_bf16(a, bq, accA[dt]);
      v16bf bt = ldB(TT, PC, dt * 16, 0, lane);
      accB[dt] = wmma_bf16(a, bt, accB[dt]);
    }
  }

  // ---- Phase 3: out = [c, A, c*A, c*Bm] ----
  {
    int row = t >> 1, g = t & 1;
    const float* cp = c + ((size_t)b * LCn + i0 + row) * Dn + g * 64;
    float* op = out + ((size_t)b * LCn + i0 + row) * (4 * Dn) + g * 64;
#pragma unroll
    for (int k2 = 0; k2 < 64; ++k2) op[k2] = cp[k2];
  }
#pragma unroll
  for (int dt = 0; dt < 8; ++dt)
#pragma unroll
    for (int r8 = 0; r8 < 8; ++r8) {
      int m    = r8 + 8 * half;
      int irow = i0 + w * 16 + m;
      int d    = dt * 16 + n;
      float cv = c[((size_t)b * LCn + irow) * Dn + d];
      float av = accA[dt][r8];
      float bv = accB[dt][r8];
      float* op = out + ((size_t)b * LCn + irow) * (4 * Dn);
      op[Dn + d]     = av;
      op[2 * Dn + d] = cv * av;
      op[3 * Dn + d] = cv * bv;
    }
}

extern "C" void kernel_launch(void* const* d_in, const int* in_sizes, int n_in,
                              void* d_out, int out_size, void* d_ws, size_t ws_size,
                              hipStream_t stream) {
  const float* c  = (const float*)d_in[0];
  const float* q  = (const float*)d_in[1];
  const float* W0 = (const float*)d_in[2];
  float* out  = (float*)d_out;
  float* T_ws = (float*)d_ws;   // B*LQ*D f32 = 4 MB

  dim3 g1(LQn / 64, Bn);
  cqa_colpass<<<g1, dim3(256), 0, stream>>>(c, q, W0, T_ws);
  dim3 g2(LCn / 64, Bn);
  cqa_rowpass<<<g2, dim3(128), 0, stream>>>(c, q, W0, T_ws, out);
}
